// AttentionHead_52596169507189
// MI455X (gfx1250) — matrix operand order repaired and checked by reading, not compiled, not run on hardware
//
#include <hip/hip_runtime.h>
#include <hip/hip_bf16.h>
#include <math.h>

// ---------------------------------------------------------------------------
// AttentionHead for MI455X (gfx1250): bf16 WMMA (v_wmma_f32_16x16x32_bf16),
// fp32 accumulate, online softmax, causal mask. Outputs (res, attn) fp32.
// B=8, T=2048, C=1024, H=64.
//
// CDNA5 paths used: v_wmma_f32_16x16x32_bf16, async global->LDS staging
// (ASYNCcnt) for weight slices (double-buffered), ds atomics for the
// cross-wave O reduction, wave32 shfl reductions for softmax.
// ---------------------------------------------------------------------------

#define BATCH 8
#define TLEN  2048
#define CDIM  1024
#define HDIM  64

typedef __attribute__((ext_vector_type(16))) __bf16 v16bf;
typedef __attribute__((ext_vector_type(8)))  __bf16 v8bf;
typedef __attribute__((ext_vector_type(8)))  float  v8f;
typedef __attribute__((ext_vector_type(4)))  float  v4f;
typedef int v4i_vs __attribute__((vector_size(16)));   // int4 (matches builtin)

#if defined(__AMDGCN__) && __has_builtin(__builtin_amdgcn_global_load_async_to_lds_b128) && \
    __has_builtin(__builtin_amdgcn_s_wait_asynccnt)
#define USE_ASYNC_LDS 1
typedef __attribute__((address_space(1))) v4i_vs* as1_v4i_ptr;
typedef __attribute__((address_space(3))) v4i_vs* as3_v4i_ptr;
#else
#define USE_ASYNC_LDS 0
#endif

__device__ __forceinline__ v8f wmma_bf16(v16bf a, v16bf b, v8f c) {
  // (neg_a, A, neg_b, B, c_mod, C, reuse_a, reuse_b)
  return __builtin_amdgcn_wmma_f32_16x16x32_bf16(false, a, false, b,
                                                 (short)0, c, false, false);
}

// A operand: 16x32 bf16 tile from row-major bf16 source (stride ld elems).
// Lane L<16: halves = K[0..7],K[16..23] of row L; lanes 16..31: K[8..15],K[24..31].
__device__ __forceinline__ v16bf load_a16_bf16(const __bf16* base, int ld, int lane) {
  const int row  = lane & 15;
  const int koff = (lane >> 4) << 3;              // 0 or 8
  const __bf16* p = base + (size_t)row * ld + koff;
  v8bf lo = *(const v8bf*)(p);
  v8bf hi = *(const v8bf*)(p + 16);
  v16bf a;
#pragma unroll
  for (int i = 0; i < 8; ++i) { a[i] = lo[i]; a[8 + i] = hi[i]; }
  return a;
}

// Same A layout, source fp32 converted to bf16 in registers.
__device__ __forceinline__ v16bf load_a16_f32(const float* base, int ld, int lane) {
  const int row  = lane & 15;
  const int koff = (lane >> 4) << 3;
  const float* p = base + (size_t)row * ld + koff;
  v4f f0 = *(const v4f*)(p);
  v4f f1 = *(const v4f*)(p + 4);
  v4f f2 = *(const v4f*)(p + 16);
  v4f f3 = *(const v4f*)(p + 20);
  v16bf a;
#pragma unroll
  for (int i = 0; i < 4; ++i) {
    a[i]      = (__bf16)f0[i];
    a[4 + i]  = (__bf16)f1[i];
    a[8 + i]  = (__bf16)f2[i];
    a[12 + i] = (__bf16)f3[i];
  }
  return a;
}

// B operand: 32x16 bf16 tile where the source is "N-major" (row n holds the 32
// consecutive K values of output column n; stride ld).
// Lane L<16: column L, K[0..7],K[16..23]; lanes 16..31: column L-16, K[8..15],K[24..31].
__device__ __forceinline__ v16bf load_b16_nmajor(const __bf16* base, int ld, int lane) {
  const int n     = lane & 15;
  const int kbase = (lane >> 4) << 3;             // 0 or 8
  const __bf16* p = base + (size_t)n * ld + kbase;
  v8bf lo = *(const v8bf*)(p);
  v8bf hi = *(const v8bf*)(p + 16);
  v16bf b;
#pragma unroll
  for (int i = 0; i < 8; ++i) { b[i] = lo[i]; b[8 + i] = hi[i]; }
  return b;
}

#if USE_ASYNC_LDS
// Stage one 64x32 bf16 weight K-slice (N-major, stride 32) into LDS with
// async global->LDS b128 copies: 8 instructions per slice (512 B per lane-wave).
__device__ __forceinline__ void async_stage_wslice(const __bf16* Wg, int kk,
                                                   __bf16* lds, int lane) {
  const int rsub = lane >> 2;                     // 0..7
  const int cseg = (lane & 3) * 8;                // elem offset within 32-wide row
#pragma unroll
  for (int i = 0; i < 8; ++i) {
    const int row = i * 8 + rsub;                 // 0..63 (h index)
    const __bf16* g = Wg + (size_t)row * CDIM + kk + cseg;
    __bf16* l = lds + row * 32 + cseg;
    __builtin_amdgcn_global_load_async_to_lds_b128(
        (as1_v4i_ptr)g, (as3_v4i_ptr)l, 0, 0);
  }
}
#endif

// ---------------------------------------------------------------------------
// Kernel 0: convert + transpose weights: Wt[w][h][c] = bf16(W_w[c][h])
// ---------------------------------------------------------------------------
__global__ void wt_convert_kernel(const float* __restrict__ Wq,
                                  const float* __restrict__ Wk,
                                  const float* __restrict__ Wv,
                                  __bf16* __restrict__ Wt) {
  int idx = blockIdx.x * blockDim.x + threadIdx.x;   // over 3*H*C
  int w   = idx / (HDIM * CDIM);
  int rem = idx - w * (HDIM * CDIM);
  int h   = rem / CDIM;
  int c   = rem - h * CDIM;
  const float* W = (w == 0) ? Wq : (w == 1) ? Wk : Wv;
  Wt[idx] = (__bf16)W[(size_t)c * HDIM + h];
}

// ---------------------------------------------------------------------------
// Kernel 1: q/k/v projection. One wave per (16-token tile, weight).
// Weight K-slices double-buffered into LDS with async copies; A operand
// register-double-buffered so WMMA overlaps the fp32 x loads.
//   q,k stored bf16 row-major [B*T, H]; v stored bf16 transposed [B, H, T].
// ---------------------------------------------------------------------------
__global__ __launch_bounds__(32) void qkv_proj_kernel(
    const float* __restrict__ x,    // [B*T, C]
    const __bf16* __restrict__ Wt,  // [3, H, C]
    __bf16* __restrict__ qb,        // [B*T, H]
    __bf16* __restrict__ kb,        // [B*T, H]
    __bf16* __restrict__ vt) {      // [B, H, T]
#if USE_ASYNC_LDS
  __shared__ __bf16 wLds[2][64 * 32];             // 2 x 4KB double buffer
#endif
  const int lane = threadIdx.x & 31;
  const int w    = blockIdx.y;                    // 0=q, 1=k, 2=v
  const int row0 = blockIdx.x * 16;               // token tile over B*T
  const __bf16* W = Wt + (size_t)w * HDIM * CDIM;

#if USE_ASYNC_LDS
  async_stage_wslice(W, 0, &wLds[0][0], lane);
#endif
  v16bf a_cur = load_a16_f32(x + (size_t)row0 * CDIM, CDIM, lane);

  v8f acc[4] = {};
  for (int s = 0; s < 32; ++s) {
    const int kk = s * 32;
    v16bf a_nxt = {};
    if (s + 1 < 32)
      a_nxt = load_a16_f32(x + (size_t)row0 * CDIM + kk + 32, CDIM, lane);
#if USE_ASYNC_LDS
    if (s + 1 < 32) {
      async_stage_wslice(W, kk + 32, &wLds[(s + 1) & 1][0], lane);
      __builtin_amdgcn_s_wait_asynccnt(8);        // current slice landed
    } else {
      __builtin_amdgcn_s_wait_asynccnt(0);
    }
    const __bf16* wb = &wLds[s & 1][0];
#pragma unroll
    for (int n = 0; n < 4; ++n) {
      v16bf bt = load_b16_nmajor(wb + (size_t)(n * 16) * 32, 32, lane);
      acc[n] = wmma_bf16(a_cur, bt, acc[n]);
    }
#else
#pragma unroll
    for (int n = 0; n < 4; ++n) {
      v16bf bt = load_b16_nmajor(W + (size_t)(n * 16) * CDIM + kk, CDIM, lane);
      acc[n] = wmma_bf16(a_cur, bt, acc[n]);
    }
#endif
    a_cur = a_nxt;
  }

  const int col   = lane & 15;
  const int rbase = (lane >> 4) << 3;             // 0 or 8
  if (w < 2) {
    __bf16* out = (w == 0) ? qb : kb;
#pragma unroll
    for (int n = 0; n < 4; ++n)
#pragma unroll
      for (int r = 0; r < 8; ++r)
        out[(size_t)(row0 + rbase + r) * HDIM + n * 16 + col] = (__bf16)acc[n][r];
  } else {
#pragma unroll
    for (int n = 0; n < 4; ++n)
#pragma unroll
      for (int r = 0; r < 8; ++r) {
        int token = row0 + rbase + r;
        int bidx  = token >> 11;                  // / TLEN
        int t     = token & (TLEN - 1);
        vt[((size_t)bidx * HDIM + n * 16 + col) * TLEN + t] = (__bf16)acc[n][r];
      }
  }
}

// ---------------------------------------------------------------------------
// Kernel 2: causal attention. 4 waves per (batch, 16-row tile); the causal
// column range is split across waves in 32-column chunks (stride 128).
//   Pass 1: per-wave online softmax stats; combined across waves via LDS.
//   Pass 2: recompute S, write attn fp32 (masked entries write 0), bounce P
//           through per-wave 1KB LDS tile into A-operand layout, accumulate
//           partial O = P*V with WMMA; cross-wave O reduction via ds_add_f32.
//   Upper triangle of attn zeroed in-kernel (no memset pre-pass).
// ---------------------------------------------------------------------------
__global__ __launch_bounds__(128) void attn_head_kernel(
    const __bf16* __restrict__ qb,  // [B*T, H]
    const __bf16* __restrict__ kb,  // [B*T, H]
    const __bf16* __restrict__ vt,  // [B, H, T]
    float* __restrict__ res,        // [B, T, H]
    float* __restrict__ attn) {     // [B, T, T]
  __shared__ __bf16 pLds[4][16 * 32];
  __shared__ float  statsM[4][16];
  __shared__ float  statsL[4][16];
  __shared__ float  oLds[16 * 64];

  const int tid   = threadIdx.x;
  const int lane  = tid & 31;
  const int wid   = tid >> 5;                     // wave 0..3
  const int b     = blockIdx.y;
  const int t0    = blockIdx.x * 16;
  const int col   = lane & 15;
  const int rbase = (lane >> 4) << 3;
  const float scale = 0.03125f;                   // 1/sqrt(C) = 1/32

  const __bf16* qbase = qb + ((size_t)b * TLEN + t0) * HDIM;
  v16bf qa0 = load_a16_bf16(qbase, HDIM, lane);
  v16bf qa1 = load_a16_bf16(qbase + 32, HDIM, lane);
  const __bf16* kbbase = kb + (size_t)b * TLEN * HDIM;

  float m[8], l[8];
#pragma unroll
  for (int r = 0; r < 8; ++r) { m[r] = -INFINITY; l[r] = 0.0f; }

  // ---- Pass 1: per-wave online softmax statistics ----
  for (int s0 = 32 * wid; s0 <= t0; s0 += 128) {
#pragma unroll
    for (int half = 0; half < 2; ++half) {
      const int j0 = s0 + 16 * half;
      if (j0 <= t0) {
        v16bf k0 = load_b16_nmajor(kbbase + (size_t)j0 * HDIM, HDIM, lane);
        v16bf k1 = load_b16_nmajor(kbbase + (size_t)j0 * HDIM + 32, HDIM, lane);
        v8f s = {};
        s = wmma_bf16(qa0, k0, s);
        s = wmma_bf16(qa1, k1, s);
        const bool diag = (j0 == t0);
#pragma unroll
        for (int r = 0; r < 8; ++r) {
          float sv = s[r] * scale;
          if (diag && (j0 + col > t0 + rbase + r)) sv = -INFINITY;
          float tm = sv;                          // row max within 16-lane half
          tm = fmaxf(tm, __shfl_xor(tm, 1, 32));
          tm = fmaxf(tm, __shfl_xor(tm, 2, 32));
          tm = fmaxf(tm, __shfl_xor(tm, 4, 32));
          tm = fmaxf(tm, __shfl_xor(tm, 8, 32));
          float nm = fmaxf(m[r], tm);
          float e  = __expf(sv - nm);
          e += __shfl_xor(e, 1, 32);
          e += __shfl_xor(e, 2, 32);
          e += __shfl_xor(e, 4, 32);
          e += __shfl_xor(e, 8, 32);
          l[r] = l[r] * __expf(m[r] - nm) + e;
          m[r] = nm;
        }
      }
    }
  }

  // ---- Combine stats across the 4 waves ----
  if (col == 0) {
#pragma unroll
    for (int r = 0; r < 8; ++r) {
      statsM[wid][rbase + r] = m[r];
      statsL[wid][rbase + r] = l[r];
    }
  }
  __syncthreads();
  float rl[8];
#pragma unroll
  for (int r = 0; r < 8; ++r) {
    const int row = rbase + r;
    float gm = statsM[0][row];
#pragma unroll
    for (int w2 = 1; w2 < 4; ++w2) gm = fmaxf(gm, statsM[w2][row]);
    float gl = 0.0f;
#pragma unroll
    for (int w2 = 0; w2 < 4; ++w2)
      gl += statsL[w2][row] * __expf(statsM[w2][row] - gm);
    m[r]  = gm;
    rl[r] = 1.0f / gl;
  }

  // ---- Pass 2: write attn, accumulate partial O = P * V ----
  v8f o[4] = {};
  __bf16* myP = &pLds[wid][0];
  float* attnRow = attn + (size_t)b * TLEN * TLEN + (size_t)t0 * TLEN;
  for (int s0 = 32 * wid; s0 <= t0; s0 += 128) {
#pragma unroll
    for (int half = 0; half < 2; ++half) {
      const int j0 = s0 + 16 * half;
      if (j0 <= t0) {
        v16bf k0 = load_b16_nmajor(kbbase + (size_t)j0 * HDIM, HDIM, lane);
        v16bf k1 = load_b16_nmajor(kbbase + (size_t)j0 * HDIM + 32, HDIM, lane);
        v8f s = {};
        s = wmma_bf16(qa0, k0, s);
        s = wmma_bf16(qa1, k1, s);
        const bool diag = (j0 == t0);
#pragma unroll
        for (int r = 0; r < 8; ++r) {
          float sv = s[r] * scale;
          if (diag && (j0 + col > t0 + rbase + r)) sv = -INFINITY;
          float p = __expf(sv - m[r]) * rl[r];
          attnRow[(size_t)(rbase + r) * TLEN + j0 + col] = p;
          myP[(rbase + r) * 32 + 16 * half + col] = (__bf16)p;
        }
      } else {
#pragma unroll
        for (int r = 0; r < 8; ++r)
          myP[(rbase + r) * 32 + 16 + col] = (__bf16)0.0f;
      }
    }
    // Intra-wave LDS visibility (DS ops are in-order within a wave; keep the
    // compiler from reordering across the cross-lane exchange).
    __builtin_amdgcn_fence(__ATOMIC_RELEASE, "wavefront");
    __builtin_amdgcn_wave_barrier();
    __builtin_amdgcn_fence(__ATOMIC_ACQUIRE, "wavefront");
    {
      const int rowA = lane & 15;
      const int koff = (lane >> 4) << 3;
      v16bf pa;
#pragma unroll
      for (int i = 0; i < 8; ++i) {
        pa[i]     = myP[rowA * 32 + koff + i];
        pa[8 + i] = myP[rowA * 32 + 16 + koff + i];
      }
#pragma unroll
      for (int n = 0; n < 4; ++n) {
        v16bf bv = load_b16_nmajor(
            vt + ((size_t)b * HDIM + n * 16) * TLEN + s0, TLEN, lane);
        o[n] = wmma_bf16(pa, bv, o[n]);
      }
    }
    __builtin_amdgcn_wave_barrier();
  }

  // ---- Cross-wave O reduction via LDS float atomics ----
  for (int i = tid; i < 16 * 64; i += 128) oLds[i] = 0.0f;
  __syncthreads();
#pragma unroll
  for (int n = 0; n < 4; ++n)
#pragma unroll
    for (int r = 0; r < 8; ++r)
      atomicAdd(&oLds[(rbase + r) * 64 + n * 16 + col], o[n][r]);
  __syncthreads();
  for (int i = tid; i < 16 * 64; i += 128) {
    const int rr = i >> 6;
    const int hh = i & 63;
    res[((size_t)b * TLEN + t0 + rr) * HDIM + hh] = oLds[i];
  }

  // ---- Zero the strict upper triangle (cols >= t0+16) for these 16 rows ----
  const v4f z4 = {};
#pragma unroll 1
  for (int r = 0; r < 16; ++r) {
    float* rowp = attnRow + (size_t)r * TLEN;
    for (int c = t0 + 16 + 4 * tid; c < TLEN; c += 512)
      *(v4f*)(rowp + c) = z4;
  }
}

// ---------------------------------------------------------------------------
extern "C" void kernel_launch(void* const* d_in, const int* in_sizes, int n_in,
                              void* d_out, int out_size, void* d_ws, size_t ws_size,
                              hipStream_t stream) {
  const float* x  = (const float*)d_in[0];
  const float* Wq = (const float*)d_in[1];
  const float* Wk = (const float*)d_in[2];
  const float* Wv = (const float*)d_in[3];

  float* res  = (float*)d_out;                                // [B,T,H]
  float* attn = res + (size_t)BATCH * TLEN * HDIM;            // [B,T,T]

  // Workspace (bf16): Wt[3,H,C] | qb[B*T,H] | kb[B*T,H] | vt[B,H,T]  (~6.7MB)
  __bf16* Wt  = (__bf16*)d_ws;
  __bf16* qb  = Wt + (size_t)3 * HDIM * CDIM;
  __bf16* kbp = qb + (size_t)BATCH * TLEN * HDIM;
  __bf16* vtp = kbp + (size_t)BATCH * TLEN * HDIM;

  wt_convert_kernel<<<dim3((3 * HDIM * CDIM) / 256), dim3(256), 0, stream>>>(
      Wq, Wk, Wv, Wt);

  qkv_proj_kernel<<<dim3((BATCH * TLEN) / 16, 3), dim3(32), 0, stream>>>(
      x, Wt, qb, kbp, vtp);

  attn_head_kernel<<<dim3(TLEN / 16, BATCH), dim3(128), 0, stream>>>(
      qb, kbp, vtp, res, attn);
}
